// CharRNN_86792699118083
// MI455X (gfx1250) — matrix operand order repaired and checked by reading, not compiled, not run on hardware
//
#include <hip/hip_runtime.h>

// CharRNN forward for MI455X (gfx1250, wave32, WMMA).
//  - prefactor  P = embedding @ W_e + b_h  (input projection becomes a gather)
//  - persistent kernel: 1024 recurrence steps, grid-wide counter barrier,
//    bf16 WMMA (v_wmma_f32_16x16x32_bf16) with f32 accumulation
//  - logits_{t-1} fused into step t (reuses the same packed h A-tiles);
//    all 32 WGs compute a 16-wide vocab tile, each stores its 8-col half
//  - weights pre-swizzled into wave32 WMMA-B layout (32B contiguous per lane per tile)
//  - h_t repacked into WMMA-A layout through LDS (s_wait_dscnt 0)
//  - K-loop fully unrolled, 3-slot register rotation = prefetch distance 2

namespace {
constexpr int Bm = 128;   // batch
constexpr int Ls = 1024;  // seq len
constexpr int Es = 512;   // embed
constexpr int Hs = 1024;  // hidden
constexpr int Vs = 256;   // vocab
constexpr int GRID  = 32; // persistent workgroups (must be co-resident)
constexpr int WAVES = 8;  // waves per WG (256 threads, wave32)

// workspace layout (bytes)
constexpr size_t OFF_P   = 0;                                    // P f32 (1 MB)
constexpr size_t OFF_WHP = OFF_P   + (size_t)Vs * Hs * 4;        // W_h bf16 packed (2 MB)
constexpr size_t OFF_WOP = OFF_WHP + (size_t)Hs * Hs * 2;        // W_out bf16 packed (0.5 MB)
constexpr size_t OFF_H0  = OFF_WOP + (size_t)Hs * Vs * 2;        // h buf 0 (0.25 MB)
constexpr size_t OFF_H1  = OFF_H0  + (size_t)Bm * Hs * 2;        // h buf 1
constexpr size_t OFF_CTR = OFF_H1  + (size_t)Bm * Hs * 2;        // barrier counter
}

typedef __attribute__((ext_vector_type(16))) __bf16 v16bf;
typedef __attribute__((ext_vector_type(8)))  float  v8f;

__device__ __forceinline__ unsigned int f2bf(float f) {
  unsigned int u = __builtin_bit_cast(unsigned int, f);
  u += 0x7FFFu + ((u >> 16) & 1u);   // round to nearest even
  return u >> 16;
}

// ---------------- prep kernels -------------------------------------------

__global__ void prep_P(const float* __restrict__ emb, const float* __restrict__ W_e,
                       const float* __restrict__ b_h, float* __restrict__ P) {
  int pid = blockIdx.x * blockDim.x + threadIdx.x;
  if (pid >= Vs * Hs) return;
  int v   = pid >> 10;
  int col = pid & (Hs - 1);
  float acc = b_h[col];
  for (int e = 0; e < Es; ++e)
    acc = fmaf(emb[v * Es + e], W_e[e * Hs + col], acc);
  P[pid] = acc;
}

// f32 [K][N] -> wave32 WMMA-B bf16 tiles (32x16). Tile (nt,kt) at uint
// offset (nt*KT+kt)*256; lane l: n = nt*16+(l&15),
// k = kt*32 + (v>=4?16:0) + (l>=16?8:0) + 2*(v&3).
__global__ void pack_B(const float* __restrict__ W, unsigned int* __restrict__ out,
                       int K, int N) {
  int gid = blockIdx.x * blockDim.x + threadIdx.x;
  if (gid >= (K * N) >> 1) return;
  int tile = gid >> 8, lane = (gid >> 3) & 31, v = gid & 7;
  int KT = K >> 5;
  int nt = tile / KT, kt = tile - nt * KT;
  int n = nt * 16 + (lane & 15);
  int k = kt * 32 + ((v >= 4) ? 16 : 0) + ((lane >= 16) ? 8 : 0) + 2 * (v & 3);
  out[gid] = f2bf(W[k * N + n]) | (f2bf(W[(k + 1) * N + n]) << 16);
}

// f32 hidden [128][1024] -> wave32 WMMA-A bf16 tiles (16x32). Tile (mt,kt) at
// uint offset (mt*32+kt)*256; lane l: m = mt*16+(l&15),
// k = kt*32 + (v>=4?16:0) + (l>=16?8:0) + 2*(v&3).
__global__ void pack_A_init(const float* __restrict__ Hd, unsigned int* __restrict__ out,
                            unsigned int* __restrict__ ctr) {
  int gid = blockIdx.x * blockDim.x + threadIdx.x;
  if (gid == 0) *ctr = 0u;                       // reset global barrier counter
  if (gid >= (Bm * Hs) >> 1) return;
  int tile = gid >> 8, lane = (gid >> 3) & 31, v = gid & 7;
  int mt = tile >> 5, kt = tile & 31;
  int m = mt * 16 + (lane & 15);
  int k = kt * 32 + ((v >= 4) ? 16 : 0) + ((lane >= 16) ? 8 : 0) + 2 * (v & 3);
  out[gid] = f2bf(Hd[m * Hs + k]) | (f2bf(Hd[m * Hs + k + 1]) << 16);
}

// ---------------- pipelined K-loop ---------------------------------------

__device__ __forceinline__ v8f wmma_bf16(v16bf a, v16bf b, v8f c) {
  return __builtin_amdgcn_wmma_f32_16x16x32_bf16(false, a, false, b, (short)0,
                                                 c, false, false);
}

// Ap/Bh0/Bh1/Bo pre-offset per (wave, wg, lane); kt tile stride = 32 elems.
// 3-slot register rotation: while computing tile kt, tile kt+1 is resident and
// tile kt+2 is being fetched into the third (dead) slot -> ~2 tile-sets of
// b128 loads in flight, hiding L2 latency behind the WMMA stream.
template <bool DO_REC, bool DO_LOG>
__device__ __forceinline__ void kloop(const v16bf* __restrict__ Ap,
                                      const v16bf* __restrict__ Bh0,
                                      const v16bf* __restrict__ Bh1,
                                      const v16bf* __restrict__ Bo,
                                      v8f& c0, v8f& c1, v8f& cl) {
  v16bf a[3], bA[3], bB[3], bO[3];
#pragma unroll
  for (int s = 0; s < 2; ++s) {          // preload tiles 0 and 1
    a[s] = Ap[s * 32];
    if (DO_REC) { bA[s] = Bh0[s * 32]; bB[s] = Bh1[s * 32]; }
    if (DO_LOG) { bO[s] = Bo[s * 32]; }
  }
#pragma unroll
  for (int kt = 0; kt < 32; ++kt) {
    const int cs = kt % 3;               // compute slot
    const int ps = (kt + 2) % 3;         // prefetch slot (disjoint from cs, cs+1)
    if (kt + 2 < 32) {
      a[ps] = Ap[(kt + 2) * 32];
      if (DO_REC) { bA[ps] = Bh0[(kt + 2) * 32]; bB[ps] = Bh1[(kt + 2) * 32]; }
      if (DO_LOG) { bO[ps] = Bo[(kt + 2) * 32]; }
    }
    if (DO_REC) {
      c0 = wmma_bf16(a[cs], bA[cs], c0);
      c1 = wmma_bf16(a[cs], bB[cs], c1);
    }
    if (DO_LOG) cl = wmma_bf16(a[cs], bO[cs], cl);
  }
}

// ---------------- persistent recurrence kernel ---------------------------

__global__ __launch_bounds__(256, 1)
void rnn_persistent(const int* __restrict__ x,        // [128][1024]
                    const float* __restrict__ P,      // [256][1024]
                    const v16bf* __restrict__ Whp,    // B-pack K=1024,N=1024
                    const v16bf* __restrict__ Wop,    // B-pack K=1024,N=256
                    unsigned int* h0u, unsigned int* h1u,
                    const float* __restrict__ b_out,
                    unsigned int* ctr,
                    float* __restrict__ out) {
  const int wg    = blockIdx.x;          // 0..31; hidden cols [32*wg, 32*wg+32)
  const int wave  = threadIdx.x >> 5;    // 0..7 ; batch rows  [16*wave, 16*wave+16)
  const int lane  = threadIdx.x & 31;
  const int lhalf = lane & 15;
  const bool hiHalf = lane >= 16;
  const int kA_hi = hiHalf ? 8 : 0;
  const int vt    = wg >> 1;             // vocab tile (16 cols), shared by WG pairs
  const bool storeLog = ((lhalf >> 3) == (wg & 1));  // my 8-col half of the tile

  const v16bf* h0r = (const v16bf*)h0u;
  const v16bf* h1r = (const v16bf*)h1u;
  uint4*       h0w = (uint4*)h0u;
  uint4*       h1w = (uint4*)h1u;

  // per-(wave,wg,lane) base pointers; kt tile stride = 32 v16bf (1 KB)
  const v16bf* Bh0 = Whp + (size_t)(wg * 2)     * 1024 + lane;
  const v16bf* Bh1 = Whp + (size_t)(wg * 2 + 1) * 1024 + lane;
  const v16bf* Bo  = Wop + (size_t)vt * 1024 + lane;

  float* logits = out;
  float* finalh = out + (size_t)Bm * Ls * Vs;

  __shared__ unsigned short lds[WAVES][16][32];   // per-wave repack staging

  for (int t = 0; t <= Ls; ++t) {
    const bool do_rec = (t < Ls);
    const bool do_log = (t > 0);
    const v16bf* Apack = (t & 1) ? h1r : h0r;     // h_{t-1}
    uint4*       Aout  = (t & 1) ? h0w : h1w;     // h_t
    const v16bf* Ap = Apack + (size_t)wave * 1024 + lane;

    v8f c0 = {}, c1 = {}, cl = {};

    if (do_rec) {
      // accumulator init = P[x[b,t], col]  (prefactored input proj + b_h)
      int xrow[8];
#pragma unroll
      for (int r = 0; r < 8; ++r) {
        int mg = wave * 16 + (hiHalf ? 8 + r : r);
        xrow[r] = x[mg * Ls + t];
      }
#pragma unroll
      for (int r = 0; r < 8; ++r) {
        c0[r] = P[xrow[r] * Hs + wg * 32 + lhalf];
        c1[r] = P[xrow[r] * Hs + wg * 32 + 16 + lhalf];
      }
    }
    if (do_log) {
      float bo = b_out[vt * 16 + lhalf];
#pragma unroll
      for (int r = 0; r < 8; ++r) cl[r] = bo;
    }

    // dispatch once per timestep -> no uniform branches in the K-loop
    if (do_rec && do_log)      kloop<true, true >(Ap, Bh0, Bh1, Bo, c0, c1, cl);
    else if (do_rec)           kloop<true, false>(Ap, Bh0, Bh1, Bo, c0, c1, cl);
    else                       kloop<false, true>(Ap, Bh0, Bh1, Bo, c0, c1, cl);

    if (do_rec) {
      // tanh, stage into LDS (m,n) order; final hidden f32 at last step
#pragma unroll
      for (int r = 0; r < 8; ++r) {
        int m = hiHalf ? 8 + r : r;
        float v0 = tanhf(c0[r]);
        float v1 = tanhf(c1[r]);
        lds[wave][m][lhalf]      = (unsigned short)f2bf(v0);
        lds[wave][m][16 + lhalf] = (unsigned short)f2bf(v1);
        if (t == Ls - 1) {
          int mg = wave * 16 + m;
          finalh[mg * Hs + wg * 32 + lhalf]      = v0;
          finalh[mg * Hs + wg * 32 + 16 + lhalf] = v1;
        }
      }
      asm volatile("s_wait_dscnt 0" ::: "memory");  // wave-local LDS ordering
      // repack this wave's 16x32 tile into WMMA-A layout (tile mt=wave, kt=wg)
      unsigned int uu[8];
#pragma unroll
      for (int v = 0; v < 8; ++v) {
        int k0 = ((v >= 4) ? 16 : 0) + kA_hi + 2 * (v & 3);
        uu[v] = *reinterpret_cast<const unsigned int*>(&lds[wave][lhalf][k0]);
      }
      uint4* dst = Aout + ((wave * 32 + wg) * 32 + lane) * 2;
      dst[0] = make_uint4(uu[0], uu[1], uu[2], uu[3]);
      dst[1] = make_uint4(uu[4], uu[5], uu[6], uu[7]);
    }

    if (do_log && storeLog) {
      // logits_{t-1}[m, vt*16 + lhalf] — this WG stores its 8-col half
#pragma unroll
      for (int r = 0; r < 8; ++r) {
        int mg = wave * 16 + (hiHalf ? 8 + r : r);
        logits[((size_t)mg * Ls + (t - 1)) * Vs + vt * 16 + lhalf] = cl[r];
      }
    }

    if (t < Ls) {
      // grid-wide barrier: release stores, arrive, spin, acquire
      __threadfence();
      __syncthreads();
      if (threadIdx.x == 0) {
        atomicAdd(ctr, 1u);
        unsigned int target = (unsigned int)(t + 1) * GRID;
        while (__hip_atomic_load(ctr, __ATOMIC_RELAXED, __HIP_MEMORY_SCOPE_AGENT) < target)
          __builtin_amdgcn_s_sleep(2);
        __threadfence();
      }
      __syncthreads();
    }
  }
}

// ---------------- host launcher ------------------------------------------

extern "C" void kernel_launch(void* const* d_in, const int* in_sizes, int n_in,
                              void* d_out, int out_size, void* d_ws, size_t ws_size,
                              hipStream_t stream) {
  const int*   x   = (const int*)  d_in[0];
  const float* hid = (const float*)d_in[1];
  const float* emb = (const float*)d_in[2];
  const float* W_e = (const float*)d_in[3];
  const float* W_h = (const float*)d_in[4];
  const float* b_h = (const float*)d_in[5];
  const float* W_o = (const float*)d_in[6];
  const float* b_o = (const float*)d_in[7];

  char* ws = (char*)d_ws;
  float*        P   = (float*)       (ws + OFF_P);
  unsigned int* Whp = (unsigned int*)(ws + OFF_WHP);
  unsigned int* Wop = (unsigned int*)(ws + OFF_WOP);
  unsigned int* h0  = (unsigned int*)(ws + OFF_H0);
  unsigned int* h1  = (unsigned int*)(ws + OFF_H1);
  unsigned int* ctr = (unsigned int*)(ws + OFF_CTR);

  prep_P<<<(Vs * Hs + 255) / 256, 256, 0, stream>>>(emb, W_e, b_h, P);
  pack_B<<<(Hs * Hs / 2 + 255) / 256, 256, 0, stream>>>(W_h, Whp, Hs, Hs);
  pack_B<<<(Hs * Vs / 2 + 255) / 256, 256, 0, stream>>>(W_o, Wop, Hs, Vs);
  pack_A_init<<<(Bm * Hs / 2 + 255) / 256, 256, 0, stream>>>(hid, h0, ctr);

  rnn_persistent<<<GRID, 256, 0, stream>>>(
      x, P, (const v16bf*)Whp, (const v16bf*)Wop, h0, h1, b_o, ctr, (float*)d_out);
}
